// EdgewiseMSA_77850577207522
// MI455X (gfx1250) — compile-verified, matrix operand order checked by reading, not run on hardware
//
#include <hip/hip_runtime.h>
#include <hip/hip_bf16.h>

typedef _Float16 half_t;
typedef __attribute__((ext_vector_type(16))) _Float16 v16h;
typedef __attribute__((ext_vector_type(8)))  float    v8f;
typedef __attribute__((ext_vector_type(4)))  unsigned int u32x4;
typedef __attribute__((ext_vector_type(8)))  unsigned int u32x8;

#define LDS_PAD 8
#define BK 64
#define LDW (BK + LDS_PAD)

// ---------------------------------------------------------------------------
// WMMA helper (CDNA5: D = A(16x32 f16) * B(32x16 f16) + C(16x16 f32))
// ---------------------------------------------------------------------------
__device__ __forceinline__ v8f wmma32(v16h a, v16h b, v8f c) {
    return __builtin_amdgcn_wmma_f32_16x16x32_f16(
        /*neg_a=*/false, a, /*neg_b=*/false, b,
        /*c_mod=*/(short)0, c, /*reuse_a=*/false, /*reuse_b=*/false);
}

// 16-bit A-matrix 16x32 layout (ISA 7.12.2): lane L holds row (L&15);
// lanes 0-15 carry K={0..7,16..23}, lanes 16-31 carry K={8..15,24..31}.
__device__ __forceinline__ v16h frag_a(const half_t* t, int ld, int lane) {
    const int row = lane & 15;
    const int kb  = (lane & 16) ? 8 : 0;
    v16h f;
#pragma unroll
    for (int i = 0; i < 16; ++i) {
        const int k = ((i >> 3) << 4) | (kb + (i & 7));
        f[i] = t[row * ld + k];
    }
    return f;
}

// B fragment from a row-major KxN tile (column `col` striped like A's rows).
__device__ __forceinline__ v16h frag_b_row(const half_t* t, int ld, int lane) {
    const int col = lane & 15;
    const int kb  = (lane & 16) ? 8 : 0;
    v16h f;
#pragma unroll
    for (int i = 0; i < 16; ++i) {
        const int k = ((i >> 3) << 4) | (kb + (i & 7));
        f[i] = t[k * ld + col];
    }
    return f;
}

// C/D 16x16 f32 layout: lane L holds column (L&15); VGPR r holds row r (+8 for lanes>=16)
__device__ __forceinline__ void store_c(float* C, int ldc, int lane, const v8f& acc) {
    const int col = lane & 15;
    const int rb  = (lane & 16) ? 8 : 0;
#pragma unroll
    for (int r = 0; r < 8; ++r) C[(long long)(rb + r) * ldc + col] = acc[r];
}

// ---------------------------------------------------------------------------
// CDNA5 async global->LDS DMA (ASYNCcnt path). Copies 64B: 4 x b128, the
// instruction offset applies to both the LDS and the memory address.
// ---------------------------------------------------------------------------
__device__ __forceinline__ void async_row64(const half_t* g, const half_t* l) {
    const unsigned loff = (unsigned)(unsigned long long)l;
    asm volatile(
        "global_load_async_to_lds_b128 %0, %1, off\n\t"
        "global_load_async_to_lds_b128 %0, %1, off offset:16\n\t"
        "global_load_async_to_lds_b128 %0, %1, off offset:32\n\t"
        "global_load_async_to_lds_b128 %0, %1, off offset:48"
        :: "v"(loff), "v"(g) : "memory");
}
__device__ __forceinline__ void wait_async0() {
    asm volatile("s_wait_asynccnt 0x0" ::: "memory");
}

// ---------------------------------------------------------------------------
// CDNA5 Tensor Data Mover: one instruction DMAs a 64x64 f16 tile (row stride
// ld_elems) from global into LDS, inserting 16B of padding every 128B so the
// LDS layout matches the [64][LDW] fragment-read layout. 2D descriptor:
// groups 2/3 are NULL.  Tracked by TENSORcnt.
// ---------------------------------------------------------------------------
__device__ __forceinline__ void tdm_load_tile64(const half_t* gtile,
                                                unsigned lds_off,
                                                unsigned ld_elems)
{
    const unsigned long long ga = (unsigned long long)gtile;
    u32x4 g0;
    g0[0] = 1u;                                   // count=1, user mode, no gather
    g0[1] = lds_off;                              // lds_addr (bytes)
    g0[2] = (unsigned)ga;                         // global_addr[31:0]
    g0[3] = (unsigned)(ga >> 32) | (2u << 30);    // global_addr[56:32] | type=2
    u32x8 g1;
    g1[0] = (1u << 16)                            // data_size = 2 bytes
          | (1u << 20)                            // pad_enable
          | (4u << 22)                            // pad_interval: every 32 DWORDs (128B)
          | (3u << 25);                           // pad_amount: 4 DWORDs (16B)
    g1[1] = (ld_elems & 0xFFFFu) << 16;           // tensor_dim0[15:0]
    g1[2] = (ld_elems >> 16) | (0xFFFFu << 16);   // tensor_dim0[31:16] | tensor_dim1 lo
    g1[3] = (64u << 16);                          // tensor_dim1 hi=0 | tile_dim0=64
    g1[4] = 64u;                                  // tile_dim1=64, tile_dim2=0
    g1[5] = ld_elems;                             // tensor_dim0_stride[31:0]
    g1[6] = 0u;                                   // stride hi / dim1_stride lo
    g1[7] = 0u;
    asm volatile("tensor_load_to_lds %0, %1" :: "s"(g0), "s"(g1) : "memory");
}

// ---------------------------------------------------------------------------
// Batched GEMM, C(MxN,f32) = A(MxK,f16) * B(NxK,f16)^T.  64x64 block tile,
// BK=64, double-buffered TDM pipeline (wave 0 drives the DMA engine),
// 4 waves in a 2x2 subtile grid.
// ---------------------------------------------------------------------------
__global__ void __launch_bounds__(128)
wmma_gemm_nt(const half_t* __restrict__ A, const half_t* __restrict__ B,
             float* __restrict__ C, int M, int N, int K,
             long long strideA, long long strideB, long long strideC)
{
    __shared__ __align__(16) half_t sA[2][64][LDW];
    __shared__ __align__(16) half_t sB[2][64][LDW];
    const long long bat = blockIdx.z;
    A += bat * strideA; B += bat * strideB; C += bat * strideC;

    const int tid  = threadIdx.x;
    const int lane = tid & 31;
    const int wid  = tid >> 5;
    const int wr   = wid >> 1, wc = wid & 1;
    const int m0   = blockIdx.x * 64;
    const int n0   = blockIdx.y * 64;

    const half_t* gA = A + (long long)m0 * K;     // tile origin rows
    const half_t* gB = B + (long long)n0 * K;
    const unsigned ldsA0 = (unsigned)(unsigned long long)&sA[0][0][0];
    const unsigned ldsB0 = (unsigned)(unsigned long long)&sB[0][0][0];
    const unsigned bufBytes = (unsigned)(sizeof(half_t) * 64 * LDW);

    v8f acc[2][2] = {};

    if (wid == 0) {
        tdm_load_tile64(gA, ldsA0, (unsigned)K);
        tdm_load_tile64(gB, ldsB0, (unsigned)K);
        __builtin_amdgcn_s_wait_tensorcnt(0);
    }
    __syncthreads();

    int buf = 0;
    for (int k0 = 0; k0 < K; k0 += BK) {
        if (k0 + BK < K && wid == 0) {            // DMA next tile into other buffer
            const unsigned nb = (buf ^ 1) ? bufBytes : 0u;
            tdm_load_tile64(gA + k0 + BK, ldsA0 + nb, (unsigned)K);
            tdm_load_tile64(gB + k0 + BK, ldsB0 + nb, (unsigned)K);
        }
#pragma unroll
        for (int ks = 0; ks < BK; ks += 32) {
            v16h a0 = frag_a(&sA[buf][wr * 32][ks],      LDW, lane);
            v16h a1 = frag_a(&sA[buf][wr * 32 + 16][ks], LDW, lane);
            v16h b0 = frag_a(&sB[buf][wc * 32][ks],      LDW, lane);
            v16h b1 = frag_a(&sB[buf][wc * 32 + 16][ks], LDW, lane);
            acc[0][0] = wmma32(a0, b0, acc[0][0]);
            acc[0][1] = wmma32(a0, b1, acc[0][1]);
            acc[1][0] = wmma32(a1, b0, acc[1][0]);
            acc[1][1] = wmma32(a1, b1, acc[1][1]);
        }
        if (wid == 0) __builtin_amdgcn_s_wait_tensorcnt(0);
        __syncthreads();    // everyone consumed buf, DMA into buf^1 landed
        buf ^= 1;
    }
#pragma unroll
    for (int i = 0; i < 2; ++i)
#pragma unroll
        for (int j = 0; j < 2; ++j)
            store_c(&C[(long long)(m0 + wr * 32 + i * 16) * N + (n0 + wc * 32 + j * 16)],
                    N, lane, acc[i][j]);
}

// ---------------------------------------------------------------------------
// Batched GEMM, C(MxN,f32) = A(MxK,f16) * B(KxN,f16)   (B row-major),
// async-LDS (ASYNCcnt) double-buffered staging.
// ---------------------------------------------------------------------------
__global__ void __launch_bounds__(128)
wmma_gemm_nn(const half_t* __restrict__ A, const half_t* __restrict__ B,
             float* __restrict__ C, int M, int N, int K,
             long long strideA, long long strideB, long long strideC)
{
    __shared__ __align__(16) half_t sA[2][64][LDW];
    __shared__ __align__(16) half_t sBn[2][BK][64 + LDS_PAD];
    const long long bat = blockIdx.z;
    A += bat * strideA; B += bat * strideB; C += bat * strideC;

    const int tid  = threadIdx.x;
    const int lane = tid & 31;
    const int wid  = tid >> 5;
    const int wr   = wid >> 1, wc = wid & 1;
    const int m0   = blockIdx.x * 64;
    const int n0   = blockIdx.y * 64;
    const int row  = tid >> 1;          // 0..63 (A rows / B k-rows)
    const int off  = (tid & 1) * 32;    // halves

    const half_t* gA = A + (long long)(m0 + row) * K + off;
    const half_t* gB = B + (long long)row * N + n0 + off;   // + k0*N per step

    v8f acc[2][2] = {};

    async_row64(gA, &sA[0][row][off]);
    async_row64(gB, &sBn[0][row][off]);
    wait_async0();
    __syncthreads();

    int buf = 0;
    for (int k0 = 0; k0 < K; k0 += BK) {
        if (k0 + BK < K) {
            async_row64(gA + k0 + BK, &sA[buf ^ 1][row][off]);
            async_row64(gB + (long long)(k0 + BK) * N, &sBn[buf ^ 1][row][off]);
        }
        if (k0 + 2 * BK < K) {
            __builtin_prefetch(gA + k0 + 2 * BK, 0, 2);
            __builtin_prefetch(gB + (long long)(k0 + 2 * BK) * N, 0, 2);
        }
#pragma unroll
        for (int ks = 0; ks < BK; ks += 32) {
            v16h a0 = frag_a(&sA[buf][wr * 32][ks],      LDW, lane);
            v16h a1 = frag_a(&sA[buf][wr * 32 + 16][ks], LDW, lane);
            v16h b0 = frag_b_row(&sBn[buf][ks][wc * 32],      64 + LDS_PAD, lane);
            v16h b1 = frag_b_row(&sBn[buf][ks][wc * 32 + 16], 64 + LDS_PAD, lane);
            acc[0][0] = wmma32(a0, b0, acc[0][0]);
            acc[0][1] = wmma32(a0, b1, acc[0][1]);
            acc[1][0] = wmma32(a1, b0, acc[1][0]);
            acc[1][1] = wmma32(a1, b1, acc[1][1]);
        }
        wait_async0();
        __syncthreads();
        buf ^= 1;
    }
#pragma unroll
    for (int i = 0; i < 2; ++i)
#pragma unroll
        for (int j = 0; j < 2; ++j)
            store_c(&C[(long long)(m0 + wr * 32 + i * 16) * N + (n0 + wc * 32 + j * 16)],
                    N, lane, acc[i][j]);
}

// ---------------------------------------------------------------------------
// Elementwise f32 -> f16 convert
// ---------------------------------------------------------------------------
__global__ void cvt_f16(const float* __restrict__ in, half_t* __restrict__ out, long long n)
{
    long long i = (long long)blockIdx.x * blockDim.x + threadIdx.x;
    const long long stride = (long long)gridDim.x * blockDim.x;
    for (; i < n; i += stride) out[i] = (half_t)in[i];
}

// ---------------------------------------------------------------------------
// Split QKV GEMM outputs (2048 x 1536 per set) into per-head q/k (f16, q
// pre-scaled by 1/sqrt(dk)) and the alpha-mixed v (f16).
// ---------------------------------------------------------------------------
__global__ void qkv_pack(const float* __restrict__ O0, const float* __restrict__ O1,
                         const float* __restrict__ chain_logit,
                         half_t* __restrict__ q0, half_t* __restrict__ k0,
                         half_t* __restrict__ q1, half_t* __restrict__ k1,
                         half_t* __restrict__ vmix)
{
    const long long idx = (long long)blockIdx.x * blockDim.x + threadIdx.x;
    if (idx >= 2048LL * 1536LL) return;
    const int rowN = (int)(idx / 1536);   // b*512 + n
    const int c    = (int)(idx % 1536);
    const int t    = c >> 9;              // 0:q 1:k 2:v
    const int rem  = c & 511;
    const int h    = rem >> 6;
    const int d    = rem & 63;
    const int b    = rowN >> 9;
    const int n    = rowN & 511;
    const long long o = (((long long)(b * 8 + h) * 512) + n) * 64 + d;
    const float x0 = O0[idx], x1 = O1[idx];
    if (t == 0) {
        q0[o] = (half_t)(x0 * 0.125f);    // 1/sqrt(64)
        q1[o] = (half_t)(x1 * 0.125f);
    } else if (t == 1) {
        k0[o] = (half_t)x0;
        k1[o] = (half_t)x1;
    } else {
        const float a = 1.0f / (1.0f + __expf(-chain_logit[0]));
        vmix[o] = (half_t)((1.0f - a) * x0 + a * x1);
    }
}

// ---------------------------------------------------------------------------
// Row softmax: one block (256 thr) per row of 512; S f32 -> A f16
// ---------------------------------------------------------------------------
__global__ void __launch_bounds__(256)
softmax_rows(const float* __restrict__ S, half_t* __restrict__ A)
{
    __shared__ float red[256];
    const long long row = blockIdx.x;
    const float* s = S + row * 512;
    half_t* a = A + row * 512;
    const int t = threadIdx.x;

    const float v0 = s[t], v1 = s[t + 256];
    red[t] = fmaxf(v0, v1);
    __syncthreads();
    for (int w = 128; w > 0; w >>= 1) {
        if (t < w) red[t] = fmaxf(red[t], red[t + w]);
        __syncthreads();
    }
    const float mx = red[0];
    __syncthreads();
    const float e0 = __expf(v0 - mx), e1 = __expf(v1 - mx);
    red[t] = e0 + e1;
    __syncthreads();
    for (int w = 128; w > 0; w >>= 1) {
        if (t < w) red[t] += red[t + w];
        __syncthreads();
    }
    const float inv = 1.0f / red[0];
    a[t]       = (half_t)(e0 * inv);
    a[t + 256] = (half_t)(e1 * inv);
}

// ---------------------------------------------------------------------------
// Edge-gate kernel: one block (256 thr) per (bh, n) row; computes the 6->16->4
// gating MLP per edge, mixes the attention maps, row-normalizes, writes M f16.
// ---------------------------------------------------------------------------
__global__ void __launch_bounds__(256)
gate_mix(const float* __restrict__ S0, const float* __restrict__ S1,
         const float* __restrict__ Cf, const float* __restrict__ Cb,
         const half_t* __restrict__ A0, const half_t* __restrict__ A1,
         const float* __restrict__ c1w, const float* __restrict__ c1b,
         const float* __restrict__ c2w, const float* __restrict__ c2b,
         half_t* __restrict__ Mout)
{
    __shared__ float w1[96], b1[16], w2[64], b2[4];
    __shared__ float red[256];
    const int t = threadIdx.x;
    if (t < 96)        w1[t]       = c1w[t];
    else if (t < 112)  b1[t - 96]  = c1b[t - 96];
    else if (t < 176)  w2[t - 112] = c2w[t - 112];
    else if (t < 180)  b2[t - 176] = c2b[t - 176];
    __syncthreads();

    const long long bh = blockIdx.x >> 9;
    const int n = blockIdx.x & 511;
    const long long base = bh * (512LL * 512LL);
    const float*  s0r = S0 + base + (long long)n * 512;
    const float*  s1r = S1 + base + (long long)n * 512;
    const float*  cfr = Cf + base + (long long)n * 512;
    const float*  cbr = Cb + base + (long long)n * 512;
    const half_t* a0r = A0 + base + (long long)n * 512;
    const half_t* a1r = A1 + base + (long long)n * 512;
    half_t* mr = Mout + base + (long long)n * 512;

    float mvals[2];
    float accum = 0.0f;
#pragma unroll
    for (int u = 0; u < 2; ++u) {
        const int m = t + u * 256;
        float f[6];
        f[0] = s0r[m];
        f[1] = s1r[m];
        f[2] = S0[base + (long long)m * 512 + n];
        f[3] = S1[base + (long long)m * 512 + n];
        f[4] = __logf(cfr[m] + 1e-6f);
        f[5] = __logf(cbr[m] + 1e-6f);
        float g[4];
#pragma unroll
        for (int o2 = 0; o2 < 4; ++o2) g[o2] = b2[o2];
#pragma unroll
        for (int o = 0; o < 16; ++o) {
            float hs = b1[o];
#pragma unroll
            for (int c = 0; c < 6; ++c) hs += f[c] * w1[o * 6 + c];
            const float x3 = hs * hs * hs;
            const float gg = 0.5f * hs *
                (1.0f + tanhf(0.7978845608028654f * (hs + 0.044715f * x3)));
#pragma unroll
            for (int o2 = 0; o2 < 4; ++o2) g[o2] += gg * w2[o2 * 16 + o];
        }
#pragma unroll
        for (int o2 = 0; o2 < 4; ++o2) g[o2] = 1.0f / (1.0f + __expf(-g[o2]));
        const float a0 = (float)a0r[m], a1 = (float)a1r[m];
        const float aand = a0 * a1;
        const float aor  = a0 + a1 - aand;
        const float anot = a0 * (1.0f - 0.5f * a1);
        const float Mv = a0 + g[0] * aand + g[1] * aor + g[2] * anot + g[3] * cfr[m];
        mvals[u] = Mv;
        accum += Mv;
    }
    red[t] = accum;
    __syncthreads();
    for (int w = 128; w > 0; w >>= 1) {
        if (t < w) red[t] += red[t + w];
        __syncthreads();
    }
    const float inv = 1.0f / (red[0] + 1e-6f);
#pragma unroll
    for (int u = 0; u < 2; ++u) mr[t + u * 256] = (half_t)(mvals[u] * inv);
}

// ---------------------------------------------------------------------------
// [b,h,n,d] f32 -> [b*512+n, h*64+d] f16 (head merge for output projection)
// ---------------------------------------------------------------------------
__global__ void attn_repack(const float* __restrict__ attnF, half_t* __restrict__ attn_h)
{
    const long long idx = (long long)blockIdx.x * blockDim.x + threadIdx.x;
    if (idx >= 4LL * 8 * 512 * 64) return;
    const int d = (int)(idx & 63);
    const int n = (int)((idx >> 6) & 511);
    const int h = (int)((idx >> 15) & 7);
    const int b = (int)(idx >> 18);
    attn_h[((long long)b * 512 + n) * 512 + h * 64 + d] = (half_t)attnF[idx];
}

// ---------------------------------------------------------------------------
extern "C" void kernel_launch(void* const* d_in, const int* in_sizes, int n_in,
                              void* d_out, int out_size, void* d_ws, size_t ws_size,
                              hipStream_t stream)
{
    const float* x       = (const float*)d_in[0];
    const float* w_qkv0  = (const float*)d_in[1];
    const float* w_qkv1  = (const float*)d_in[2];
    const float* conv1_w = (const float*)d_in[3];
    const float* conv1_b = (const float*)d_in[4];
    const float* conv2_w = (const float*)d_in[5];
    const float* conv2_b = (const float*)d_in[6];
    const float* w_proj  = (const float*)d_in[7];
    const float* chain   = (const float*)d_in[8];
    float* out = (float*)d_out;

    (void)in_sizes; (void)n_in; (void)out_size; (void)ws_size;

    char* ws = (char*)d_ws;
    size_t off = 0;
    auto alloc = [&](size_t bytes) -> void* {
        void* p = ws + off;
        off += (bytes + 255) & ~(size_t)255;
        return p;
    };
    const long long BH = 32, N = 512, DK = 64, BN = 2048, D = 512, TD = 1536;

    half_t* xh    = (half_t*)alloc(BN * D * 2);
    half_t* w0h   = (half_t*)alloc(TD * D * 2);
    half_t* w1h   = (half_t*)alloc(TD * D * 2);
    half_t* wph   = (half_t*)alloc(D * D * 2);
    float*  O0    = (float*) alloc(BN * TD * 4);
    float*  O1    = (float*) alloc(BN * TD * 4);
    half_t* q0h   = (half_t*)alloc(BH * N * DK * 2);
    half_t* k0h   = (half_t*)alloc(BH * N * DK * 2);
    half_t* q1h   = (half_t*)alloc(BH * N * DK * 2);
    half_t* k1h   = (half_t*)alloc(BH * N * DK * 2);
    half_t* vmixh = (half_t*)alloc(BH * N * DK * 2);
    float*  S0    = (float*) alloc(BH * N * N * 4);
    float*  S1    = (float*) alloc(BH * N * N * 4);
    half_t* A0h   = (half_t*)alloc(BH * N * N * 2);
    half_t* A1h   = (half_t*)alloc(BH * N * N * 2);
    float*  Cf    = (float*) alloc(BH * N * N * 4);
    float*  Cb    = (float*) alloc(BH * N * N * 4);
    half_t* Mh    = (half_t*)alloc(BH * N * N * 2);
    float*  attnF = (float*) alloc(BH * N * DK * 4);
    half_t* attnH = (half_t*)alloc(BN * D * 2);

    // 1. converts
    cvt_f16<<<dim3(1024), dim3(256), 0, stream>>>(x,      xh,  BN * D);
    cvt_f16<<<dim3(1024), dim3(256), 0, stream>>>(w_qkv0, w0h, TD * D);
    cvt_f16<<<dim3(1024), dim3(256), 0, stream>>>(w_qkv1, w1h, TD * D);
    cvt_f16<<<dim3(512),  dim3(256), 0, stream>>>(w_proj, wph, D * D);

    // 2. QKV GEMMs: O = Xh (2048x512) * W^T (512x1536)
    wmma_gemm_nt<<<dim3(32, 24, 1), dim3(128), 0, stream>>>(
        xh, w0h, O0, 2048, 1536, 512, 0, 0, 0);
    wmma_gemm_nt<<<dim3(32, 24, 1), dim3(128), 0, stream>>>(
        xh, w1h, O1, 2048, 1536, 512, 0, 0, 0);

    // 3. split/pack (scale folded into q, alpha-mix folded into v)
    qkv_pack<<<dim3(12288), dim3(256), 0, stream>>>(O0, O1, chain,
                                                    q0h, k0h, q1h, k1h, vmixh);

    // 4. scores: S = q * k^T, batched over 32 (b,h)
    wmma_gemm_nt<<<dim3(8, 8, 32), dim3(128), 0, stream>>>(
        q0h, k0h, S0, 512, 512, 64, N * DK, N * DK, N * N);
    wmma_gemm_nt<<<dim3(8, 8, 32), dim3(128), 0, stream>>>(
        q1h, k1h, S1, 512, 512, 64, N * DK, N * DK, N * N);

    // 5. softmax rows
    softmax_rows<<<dim3(16384), dim3(256), 0, stream>>>(S0, A0h);
    softmax_rows<<<dim3(16384), dim3(256), 0, stream>>>(S1, A1h);

    // 6. chained maps: Cf = A0*A1, Cb = A1*A0 (batched 512^3)
    wmma_gemm_nn<<<dim3(8, 8, 32), dim3(128), 0, stream>>>(
        A0h, A1h, Cf, 512, 512, 512, N * N, N * N, N * N);
    wmma_gemm_nn<<<dim3(8, 8, 32), dim3(128), 0, stream>>>(
        A1h, A0h, Cb, 512, 512, 512, N * N, N * N, N * N);

    // 7. edge gating + row-normalized mix -> M (f16)
    gate_mix<<<dim3(16384), dim3(256), 0, stream>>>(
        S0, S1, Cf, Cb, A0h, A1h, conv1_w, conv1_b, conv2_w, conv2_b, Mh);

    // 8. attention output: attn = M * vmix (batched 512x64x512)
    wmma_gemm_nn<<<dim3(8, 1, 32), dim3(128), 0, stream>>>(
        Mh, vmixh, attnF, 512, 64, 512, N * N, N * DK, N * DK);

    // 9. head merge + final projection: out = attn (2048x512) * w_proj^T
    attn_repack<<<dim3(4096), dim3(256), 0, stream>>>(attnF, attnH);
    wmma_gemm_nt<<<dim3(32, 8, 1), dim3(128), 0, stream>>>(
        attnH, wph, out, 2048, 512, 512, 0, 0, 0);
}
